// WindowAttention_20538533609574
// MI455X (gfx1250) — matrix operand order repaired and checked by reading, not compiled, not run on hardware
//
#include <hip/hip_runtime.h>
#include <hip/hip_bf16.h>
#include <stdint.h>

typedef __attribute__((ext_vector_type(16))) __bf16 v16bf;
typedef __attribute__((ext_vector_type(8)))  float  v8f;

union ABu {
  v16bf v;
  uint4 q[2];
  unsigned short u[16];
};

__device__ __forceinline__ unsigned short f2bf(float f) {
  union { float f; unsigned u; } x; x.f = f;
  unsigned r = x.u + 0x7FFFu + ((x.u >> 16) & 1u);
  return (unsigned short)(r >> 16);
}

__device__ __forceinline__ v8f vzero8() {
  v8f z;
#pragma unroll
  for (int i = 0; i < 8; ++i) z[i] = 0.0f;
  return z;
}

// ---------------------------------------------------------------------------
// Kernel 0: precompute combined additive term
//   cmb[w][h][i][j] = bias_table[rel_index[i*49+j]][h] + mask[w][i][j]
// 64*12*2401 f32 = 7 MB, L2-resident; removes the dependent gather chain
// from the 49152 attention wave-tasks.
// ---------------------------------------------------------------------------
__global__ __launch_bounds__(256) void bias_mask_kernel(
    const float* __restrict__ bias_table, const int* __restrict__ rel_index,
    const float* __restrict__ mask, float* __restrict__ cmb)
{
  const int wh = blockIdx.x;                 // 0..767 = w*12 + h
  const int w = wh / 12, hh = wh - w * 12;
  const float* mbase = mask + (size_t)w * 2401;
  float* obase = cmb + (size_t)wh * 2401;
  for (int e = threadIdx.x; e < 2401; e += 256)
    obase[e] = bias_table[rel_index[e] * 12 + hh] + mbase[e];
}

// ---------------------------------------------------------------------------
// Kernel A: QKV projection GEMM  (200704 x 1152) = x(200704x384) @ qkv_w^T
// Tile 128x128x32, 256 threads = 8 waves (2 M x 4 N), wave tile 64x32.
// Epilogue scatters bf16 into head-major q (scaled), k, vT buffers.
// ---------------------------------------------------------------------------
__global__ __launch_bounds__(256) void qkv_gemm_kernel(
    const float* __restrict__ x, const float* __restrict__ w,
    const float* __restrict__ bias, unsigned short* __restrict__ qs,
    unsigned short* __restrict__ ks, unsigned short* __restrict__ vT)
{
  __shared__ unsigned short As[128 * 40];   // [row][k], stride 40 halves (pad)
  __shared__ unsigned short Bs[128 * 40];   // [col][k], stride 40 halves

  const int t  = threadIdx.x;
  const int bx = blockIdx.x;                // 0..8  over N=1152
  const int by = blockIdx.y;                // 0..1567 over M=200704
  const int wave = t >> 5, lane = t & 31;
  const int wm = wave >> 2, wn = wave & 3;
  const int h  = lane >> 4, nn = lane & 15;

  const int lrow = t >> 1;                  // 0..127
  const int lcol = (t & 1) << 4;            // 0 or 16
  const float* xp = x + (size_t)(by * 128 + lrow) * 384 + lcol;
  const float* wp = w + (size_t)(bx * 128 + lrow) * 384 + lcol;
  unsigned short* asp = &As[lrow * 40 + lcol];
  unsigned short* bsp = &Bs[lrow * 40 + lcol];

  v8f acc[4][2];
#pragma unroll
  for (int mt = 0; mt < 4; ++mt)
#pragma unroll
    for (int nt = 0; nt < 2; ++nt) acc[mt][nt] = vzero8();

  for (int k0 = 0; k0 < 384; k0 += 32) {
    struct alignas(16) { unsigned short u[16]; } ta, tb;
    const float4* xs = (const float4*)(xp + k0);
    const float4* wsv = (const float4*)(wp + k0);
    if (k0 + 64 < 384) {                    // prefetch 2 K-tiles ahead
      __builtin_prefetch(xp + k0 + 64, 0, 3);
      __builtin_prefetch(wp + k0 + 64, 0, 3);
    }
#pragma unroll
    for (int i = 0; i < 4; ++i) {
      float4 fa = xs[i]; float4 fb = wsv[i];
      ta.u[4*i+0] = f2bf(fa.x); ta.u[4*i+1] = f2bf(fa.y);
      ta.u[4*i+2] = f2bf(fa.z); ta.u[4*i+3] = f2bf(fa.w);
      tb.u[4*i+0] = f2bf(fb.x); tb.u[4*i+1] = f2bf(fb.y);
      tb.u[4*i+2] = f2bf(fb.z); tb.u[4*i+3] = f2bf(fb.w);
    }
    *(uint4*)asp       = *(uint4*)&ta.u[0];
    *(uint4*)(asp + 8) = *(uint4*)&ta.u[8];
    *(uint4*)bsp       = *(uint4*)&tb.u[0];
    *(uint4*)(bsp + 8) = *(uint4*)&tb.u[8];
    __syncthreads();

    ABu A[4], B;
#pragma unroll
    for (int mt = 0; mt < 4; ++mt) {
      const unsigned short* ar = &As[(wm * 64 + mt * 16 + nn) * 40 + h * 16];
      A[mt].q[0] = *(const uint4*)ar;
      A[mt].q[1] = *(const uint4*)(ar + 8);
    }
#pragma unroll
    for (int nt = 0; nt < 2; ++nt) {
      const unsigned short* br = &Bs[(wn * 32 + nt * 16 + nn) * 40 + h * 16];
      B.q[0] = *(const uint4*)br;
      B.q[1] = *(const uint4*)(br + 8);
#pragma unroll
      for (int mt = 0; mt < 4; ++mt)
        acc[mt][nt] = __builtin_amdgcn_wmma_f32_16x16x32_bf16(
            false, A[mt].v, false, B.v, (short)0, acc[mt][nt], false, false);
    }
    __syncthreads();
  }

  const int part = bx / 3;                  // 0=q 1=k 2=v (384 = 3 bx tiles)
#pragma unroll
  for (int nt = 0; nt < 2; ++nt) {
    const int gcol = bx * 128 + wn * 32 + nt * 16 + nn;
    const int rem  = gcol - part * 384;
    const int head = rem >> 5, d = rem & 31;
    const float bv = bias[gcol];
#pragma unroll
    for (int mt = 0; mt < 4; ++mt) {
#pragma unroll
      for (int r = 0; r < 8; ++r) {
        const int grow = by * 128 + wm * 64 + mt * 16 + r + 8 * h;
        const int bwin = grow / 49;
        const int tok  = grow - bwin * 49;
        const size_t bh = (size_t)bwin * 12 + head;
        const float v = acc[mt][nt][r] + bv;
        if (part == 0)
          qs[(bh * 64 + tok) * 32 + d] = f2bf(v * 0.17677669529663687f);
        else if (part == 1)
          ks[(bh * 64 + tok) * 32 + d] = f2bf(v);
        else
          vT[(bh * 32 + d) * 64 + tok] = f2bf(v);
      }
    }
  }
}

// ---------------------------------------------------------------------------
// Kernel B: fused window attention. One wave per (window, head).
// S=QK^T (4 WMMA) -> LDS -> (+cmb) softmax -> P (bf16, unnormalized)
// -> O=P*V (4 WMMA), rows scaled by 1/rowsum at store.
// ---------------------------------------------------------------------------
__global__ __launch_bounds__(128) void win_attn_kernel(
    const unsigned short* __restrict__ qs, const unsigned short* __restrict__ ks,
    const unsigned short* __restrict__ vT, const float* __restrict__ cmb,
    unsigned short* __restrict__ ao)
{
  __shared__ float          Sbuf[4][16 * 65];   // stride 65 -> no bank conflicts
  __shared__ unsigned short Pbuf[4][16 * 72];   // stride 72 halves, 16B aligned rows
  __shared__ float          Rsum[4][16];

  const int t = threadIdx.x, wave = t >> 5, lane = t & 31;
  const int h = lane >> 4, nn = lane & 15;
  const int wid = blockIdx.x * 4 + wave;        // 0..49151 = b*12 + head
  const int b = wid / 12, hh = wid - b * 12;
  const unsigned short* qp = qs + (size_t)wid * 2048;
  const unsigned short* kp = ks + (size_t)wid * 2048;
  const unsigned short* vp = vT + (size_t)wid * 2048;
  const float* cbase = cmb + ((size_t)((b & 63) * 12 + hh)) * 2401;

  ABu K[4], V[4];
#pragma unroll
  for (int ct = 0; ct < 4; ++ct) {              // B-frag of K^T: n=key token, K=d
    const unsigned short* src = kp + (ct * 16 + nn) * 32 + h * 16;
    K[ct].q[0] = *(const uint4*)src;
    K[ct].q[1] = *(const uint4*)(src + 8);
  }
#pragma unroll
  for (int kt = 0; kt < 2; ++kt)
#pragma unroll
    for (int nt = 0; nt < 2; ++nt) {            // B-frag of V: n=d, K=token
      ABu& vv = V[kt * 2 + nt];
      const int d  = nt * 16 + nn;
      const int t0 = kt * 32 + h * 16;
      const unsigned short* src = vp + d * 64 + t0;
      vv.q[0] = *(const uint4*)src;
      vv.q[1] = *(const uint4*)(src + 8);
#pragma unroll
      for (int e = 0; e < 16; ++e)              // kill padding tokens >= 49
        if (t0 + e >= 49) vv.u[e] = 0;
    }

  float* sb = Sbuf[wave];
  unsigned short* pb = Pbuf[wave];

  for (int mt = 0; mt < 4; ++mt) {
    ABu A;
    const unsigned short* qrow = qp + (mt * 16 + nn) * 32 + h * 16;
    A.q[0] = *(const uint4*)qrow;
    A.q[1] = *(const uint4*)(qrow + 8);
    v8f S[4];
#pragma unroll
    for (int ct = 0; ct < 4; ++ct)
      S[ct] = __builtin_amdgcn_wmma_f32_16x16x32_bf16(
          false, A.v, false, K[ct].v, (short)0, vzero8(), false, false);
#pragma unroll
    for (int ct = 0; ct < 4; ++ct)
#pragma unroll
      for (int r = 0; r < 8; ++r)
        sb[(r + 8 * h) * 65 + ct * 16 + nn] = S[ct][r];
    __syncthreads();

    {   // softmax: lane pair (nn, nn+16) both handle row nn (same values)
      const int row = nn;
      const int gi = mt * 16 + row;
      const int ii = gi < 49 ? gi : 48;
      const float* mrow = cbase + ii * 49;
      float* srow = &sb[row * 65];
      float mx = -1e30f;
      for (int j = 0; j < 49; ++j) {
        float s = srow[j] + mrow[j];
        srow[j] = s;
        mx = fmaxf(mx, s);
      }
      float sum = 0.0f;
      unsigned short* prow = &pb[row * 72];
      for (int j = 0; j < 49; ++j) {
        float p = __expf(srow[j] - mx);
        sum += p;
        prow[j] = f2bf(p);
      }
      for (int j = 49; j < 64; ++j) prow[j] = 0;
      Rsum[wave][row] = 1.0f / sum;
    }
    __syncthreads();

    v8f O[2]; O[0] = vzero8(); O[1] = vzero8();
#pragma unroll
    for (int kt = 0; kt < 2; ++kt) {            // A-frag of P from LDS
      ABu P;
      const unsigned short* pr = &pb[nn * 72 + kt * 32 + h * 16];
      P.q[0] = *(const uint4*)pr;
      P.q[1] = *(const uint4*)(pr + 8);
#pragma unroll
      for (int nt = 0; nt < 2; ++nt)
        O[nt] = __builtin_amdgcn_wmma_f32_16x16x32_bf16(
            false, P.v, false, V[kt * 2 + nt].v, (short)0, O[nt], false, false);
    }
    float rs[8];
#pragma unroll
    for (int r = 0; r < 8; ++r) rs[r] = Rsum[wave][r + 8 * h];
#pragma unroll
    for (int nt = 0; nt < 2; ++nt)
#pragma unroll
      for (int r = 0; r < 8; ++r) {
        const int tok = mt * 16 + r + 8 * h;
        if (tok < 49)
          ao[((size_t)b * 49 + tok) * 384 + hh * 32 + nt * 16 + nn] =
              f2bf(O[nt][r] * rs[r]);
      }
    __syncthreads();
  }
}

// ---------------------------------------------------------------------------
// Kernel C: output projection GEMM (200704 x 384) = attn_out(bf16) @ proj_w^T
// ---------------------------------------------------------------------------
__global__ __launch_bounds__(256) void proj_gemm_kernel(
    const unsigned short* __restrict__ a, const float* __restrict__ w,
    const float* __restrict__ bias, float* __restrict__ out)
{
  __shared__ unsigned short As[128 * 40];
  __shared__ unsigned short Bs[128 * 40];

  const int t  = threadIdx.x;
  const int bx = blockIdx.x;                // 0..2 over N=384
  const int by = blockIdx.y;                // 0..1567
  const int wave = t >> 5, lane = t & 31;
  const int wm = wave >> 2, wn = wave & 3;
  const int h  = lane >> 4, nn = lane & 15;

  const int lrow = t >> 1;
  const int lcol = (t & 1) << 4;
  const unsigned short* ap = a + (size_t)(by * 128 + lrow) * 384 + lcol;
  const float* wp = w + (size_t)(bx * 128 + lrow) * 384 + lcol;
  unsigned short* asp = &As[lrow * 40 + lcol];
  unsigned short* bsp = &Bs[lrow * 40 + lcol];

  v8f acc[4][2];
#pragma unroll
  for (int mt = 0; mt < 4; ++mt)
#pragma unroll
    for (int nt = 0; nt < 2; ++nt) acc[mt][nt] = vzero8();

  for (int k0 = 0; k0 < 384; k0 += 32) {
    uint4 va0 = *(const uint4*)(ap + k0);
    uint4 va1 = *(const uint4*)(ap + k0 + 8);
    struct alignas(16) { unsigned short u[16]; } tb;
    const float4* wsv = (const float4*)(wp + k0);
    if (k0 + 64 < 384) {
      __builtin_prefetch(ap + k0 + 64, 0, 3);
      __builtin_prefetch(wp + k0 + 64, 0, 3);
    }
#pragma unroll
    for (int i = 0; i < 4; ++i) {
      float4 fb = wsv[i];
      tb.u[4*i+0] = f2bf(fb.x); tb.u[4*i+1] = f2bf(fb.y);
      tb.u[4*i+2] = f2bf(fb.z); tb.u[4*i+3] = f2bf(fb.w);
    }
    *(uint4*)asp       = va0;
    *(uint4*)(asp + 8) = va1;
    *(uint4*)bsp       = *(uint4*)&tb.u[0];
    *(uint4*)(bsp + 8) = *(uint4*)&tb.u[8];
    __syncthreads();

    ABu A[4], B;
#pragma unroll
    for (int mt = 0; mt < 4; ++mt) {
      const unsigned short* ar = &As[(wm * 64 + mt * 16 + nn) * 40 + h * 16];
      A[mt].q[0] = *(const uint4*)ar;
      A[mt].q[1] = *(const uint4*)(ar + 8);
    }
#pragma unroll
    for (int nt = 0; nt < 2; ++nt) {
      const unsigned short* br = &Bs[(wn * 32 + nt * 16 + nn) * 40 + h * 16];
      B.q[0] = *(const uint4*)br;
      B.q[1] = *(const uint4*)(br + 8);
#pragma unroll
      for (int mt = 0; mt < 4; ++mt)
        acc[mt][nt] = __builtin_amdgcn_wmma_f32_16x16x32_bf16(
            false, A[mt].v, false, B.v, (short)0, acc[mt][nt], false, false);
    }
    __syncthreads();
  }

#pragma unroll
  for (int nt = 0; nt < 2; ++nt) {
    const int gcol = bx * 128 + wn * 32 + nt * 16 + nn;
    const float bv = bias[gcol];
#pragma unroll
    for (int mt = 0; mt < 4; ++mt)
#pragma unroll
      for (int r = 0; r < 8; ++r) {
        const int grow = by * 128 + wm * 64 + mt * 16 + r + 8 * h;
        out[(size_t)grow * 384 + gcol] = acc[mt][nt][r] + bv;
      }
  }
}

// ---------------------------------------------------------------------------
extern "C" void kernel_launch(void* const* d_in, const int* in_sizes, int n_in,
                              void* d_out, int out_size, void* d_ws, size_t ws_size,
                              hipStream_t stream) {
  (void)in_sizes; (void)n_in; (void)out_size; (void)ws_size;
  const float* x          = (const float*)d_in[0];
  const float* mask       = (const float*)d_in[1];
  const float* qkv_w      = (const float*)d_in[2];
  const float* qkv_b      = (const float*)d_in[3];
  const float* proj_w     = (const float*)d_in[4];
  const float* proj_b     = (const float*)d_in[5];
  const float* bias_table = (const float*)d_in[6];
  const int*   rel_index  = (const int*)d_in[7];
  float* out = (float*)d_out;

  // workspace layout:
  // q: [4096*12][64][32] bf16   k: same   vT: [4096*12][32][64] bf16
  // ao: [4096*49][384] bf16     cmb: [64][12][49][49] f32
  const size_t SZQ  = (size_t)4096 * 12 * 64 * 32 * 2;   // 201326592 B
  const size_t SZAO = (size_t)4096 * 49 * 384 * 2;       // 154140672 B
  char* ws = (char*)d_ws;
  unsigned short* qs = (unsigned short*)(ws);
  unsigned short* ks = (unsigned short*)(ws + SZQ);
  unsigned short* vT = (unsigned short*)(ws + 2 * SZQ);
  unsigned short* ao = (unsigned short*)(ws + 3 * SZQ);
  float*          cm = (float*)(ws + 3 * SZQ + SZAO);

  bias_mask_kernel<<<dim3(768), 256, 0, stream>>>(bias_table, rel_index, mask, cm);
  qkv_gemm_kernel<<<dim3(9, 1568), 256, 0, stream>>>(x, qkv_w, qkv_b, qs, ks, vT);
  win_attn_kernel<<<dim3(12288), 128, 0, stream>>>(qs, ks, vT, cm, ao);
  proj_gemm_kernel<<<dim3(3, 1568), 256, 0, stream>>>(ao, proj_w, proj_b, out);
}
